// GRU_17411797418613
// MI455X (gfx1250) — matrix operand order repaired and checked by reading, not compiled, not run on hardware
//
#include <hip/hip_runtime.h>
#include <hip/hip_bf16.h>

// ---------------------------------------------------------------------------
// GRU: input projections as WMMA bf16 GEMMs, recurrent scan as a persistent
// grid-synchronized GEMV kernel with bf16 weights resident in L2.
// ---------------------------------------------------------------------------

#define T_LEN 4096
#define HDIM  2048
#define KDIM  2048
#define NWG   64          // persistent scan workgroups (must be co-resident)
#define CHUNK (HDIM / NWG) // 32 output columns per workgroup

typedef __attribute__((ext_vector_type(16))) __bf16 v16bf;
typedef __attribute__((ext_vector_type(8)))  float  v8f;

union Frag {
    uint4 u[2];   // 32 bytes = 16 bf16
    v16bf v;
};

static __device__ __forceinline__ unsigned short f2bf(float f) {
    unsigned u = __float_as_uint(f);
    u += 0x7FFFu + ((u >> 16) & 1u);   // round-to-nearest-even
    return (unsigned short)(u >> 16);
}
static __device__ __forceinline__ float bf2f_lo(unsigned packed) {
    return __uint_as_float(packed << 16);
}
static __device__ __forceinline__ float bf2f_hi(unsigned packed) {
    return __uint_as_float(packed & 0xFFFF0000u);
}

// ---------------------------------------------------------------------------
// init: zero h ping-pong buffer and grid barrier counter (ws is poisoned 0xAA)
// ---------------------------------------------------------------------------
__global__ __launch_bounds__(256) void init_state(float* hbuf, unsigned* ctr) {
    for (int i = threadIdx.x; i < 2 * HDIM; i += 256) hbuf[i] = 0.0f;
    if (threadIdx.x == 0) *ctr = 0u;
}

// ---------------------------------------------------------------------------
// fp32 -> bf16 bitcast copy (inputs)
// ---------------------------------------------------------------------------
__global__ __launch_bounds__(256) void convert_bf16(const float* __restrict__ src,
                                                    unsigned short* __restrict__ dst,
                                                    int n) {
    int stride = gridDim.x * 256;
    for (int i = blockIdx.x * 256 + threadIdx.x; i < n; i += stride)
        dst[i] = f2bf(src[i]);
}

// ---------------------------------------------------------------------------
// LDS-tiled transpose + bf16 convert: src [n x n] fp32 row-major ->
// dst [n x n] bf16 row-major holding src^T (row j = column j of src).
// ---------------------------------------------------------------------------
__global__ __launch_bounds__(256) void transpose_bf16(const float* __restrict__ src,
                                                      unsigned short* __restrict__ dst,
                                                      int n) {
    __shared__ float tile[32][33];
    const int tx = threadIdx.x & 31;
    const int ty = threadIdx.x >> 5;          // 0..7
    const int bx = blockIdx.x * 32;
    const int by = blockIdx.y * 32;
#pragma unroll
    for (int j = 0; j < 32; j += 8)
        tile[ty + j][tx] = src[(size_t)(by + ty + j) * n + (bx + tx)];
    __syncthreads();
#pragma unroll
    for (int j = 0; j < 32; j += 8)
        dst[(size_t)(bx + ty + j) * n + (by + tx)] = f2bf(tile[tx][ty + j]);
}

// ---------------------------------------------------------------------------
// WMMA bf16 GEMM: out[T x N] = X[T x K] * W[K x N] + bias, with W supplied
// TRANSPOSED as WT[N x K] bf16 row-major (contiguous K per output column).
// One 16x16 C tile per wave, K-loop step 32 (v_wmma_f32_16x16x32_bf16).
//
// Fragment layouts per CDNA5 ISA 7.12.2 (wave32):
//  A 16x32: lanes 0-15 row M=lane, K={0..7,16..23}; lanes 16-31 same row,
//           K={8..15,24..31}  -> two contiguous 16B loads per lane.
//  B 32x16: lanes 0-15 col N=lane, K={0..7,16..23}; lanes 16-31 col, K={8..15,24..31}
//           -> two contiguous 16B loads per lane from WT.
//  C/D:     vgpr v: lanes 0-15 -> M=v, lanes 16-31 -> M=v+8; N=lane%16.
// ---------------------------------------------------------------------------
__global__ __launch_bounds__(256) void gemm_wmma_bf16(
    const unsigned short* __restrict__ X,    // [T x K] bf16
    const unsigned short* __restrict__ WT,   // [N x K] bf16 (transposed weight)
    const float* __restrict__ bias,          // [N]
    float* __restrict__ out,                 // [T x N] fp32
    int T, int K, int N) {
    const int lane = threadIdx.x & 31;
    const int wave = threadIdx.x >> 5;
    const int tilesN = N >> 4;
    const int tile = blockIdx.x * 8 + wave;
    const int tm = tile / tilesN;
    const int tn = tile % tilesN;
    if (tm >= (T >> 4)) return;

    const int half = lane >> 4;   // 0 or 1
    const int l16  = lane & 15;

    const unsigned short* aRow = X  + (size_t)(tm * 16 + l16) * K;
    const unsigned short* bRow = WT + (size_t)(tn * 16 + l16) * K;

    v8f acc = {};
    for (int k0 = 0; k0 < K; k0 += 32) {
        Frag a, b;
        a.u[0] = *(const uint4*)(aRow + k0 + half * 8);        // K +0..7   (+8 if hi half)
        a.u[1] = *(const uint4*)(aRow + k0 + 16 + half * 8);   // K +16..23 (+8 if hi half)
        b.u[0] = *(const uint4*)(bRow + k0 + half * 8);        // K +0..7   (+8 if hi half)
        b.u[1] = *(const uint4*)(bRow + k0 + 16 + half * 8);   // K +16..23 (+8 if hi half)
        acc = __builtin_amdgcn_wmma_f32_16x16x32_bf16(
            /*neg_a=*/false, a.v, /*neg_b=*/false, b.v,
            /*c_mod=*/(short)0, acc, /*reuse_a=*/false, /*reuse_b=*/false);
    }

    const int col = tn * 16 + l16;
    const float bv = bias[col];
#pragma unroll
    for (int v = 0; v < 8; ++v) {
        const int row = tm * 16 + v + half * 8;
        out[(size_t)row * N + col] = acc[v] + bv;
    }
}

// ---------------------------------------------------------------------------
// Persistent GRU scan. Each block owns CHUNK output columns; recurrent
// weights are bf16, transposed (column-contiguous), streamed from L2 each
// step. h is double-buffered in global; one grid barrier per step via a
// monotonic atomic counter (NWG blocks assumed co-resident).
// 8 threads cooperate per column (shfl_xor reduction within lane-octets).
// ---------------------------------------------------------------------------
__global__ __launch_bounds__(256) void gru_scan(
    const unsigned short* __restrict__ WThr,  // [H x H] bf16, row j = col j of W_hr
    const unsigned short* __restrict__ WThz,
    const unsigned short* __restrict__ WThn,
    const float* __restrict__ b_hn,
    const float* __restrict__ xr,             // [T x H]
    const float* __restrict__ xz,             // [T x H]
    float* __restrict__ xn_io,                // d_out: holds xn, overwritten with h
    float* __restrict__ hbuf,                 // [2 x H] ping-pong
    unsigned* __restrict__ ctr) {
    __shared__ float sh[HDIM];
    const int tid = threadIdx.x;
    const int col = blockIdx.x * CHUNK + (tid >> 3);  // 32 cols per block
    const int sub = tid & 7;                          // 8 threads per column

    const unsigned short* wr = WThr + (size_t)col * HDIM;
    const unsigned short* wz = WThz + (size_t)col * HDIM;
    const unsigned short* wn = WThn + (size_t)col * HDIM;
    const float bn = b_hn[col];

    for (int t = 0; t < T_LEN; ++t) {
        const float* hin  = hbuf + ((t & 1) ? HDIM : 0);
        float*       hout = hbuf + ((t & 1) ? 0 : HDIM);

        // snapshot h into LDS
        for (int i = tid; i < HDIM; i += 256) sh[i] = hin[i];
        __syncthreads();

        float dr = 0.f, dz = 0.f, dn = 0.f;
#pragma unroll 4
        for (int it = 0; it < HDIM / 64; ++it) {      // 32 iterations
            const int kb = sub * 8 + it * 64;         // 8 contiguous bf16 per load
            const uint4 ur = *(const uint4*)(wr + kb);
            const uint4 uz = *(const uint4*)(wz + kb);
            const uint4 un = *(const uint4*)(wn + kb);
#pragma unroll
            for (int e = 0; e < 4; ++e) {
                const unsigned pr = ((const unsigned*)&ur)[e];
                const unsigned pz = ((const unsigned*)&uz)[e];
                const unsigned pn = ((const unsigned*)&un)[e];
                const float h0 = sh[kb + 2 * e];
                const float h1 = sh[kb + 2 * e + 1];
                dr += bf2f_lo(pr) * h0 + bf2f_hi(pr) * h1;
                dz += bf2f_lo(pz) * h0 + bf2f_hi(pz) * h1;
                dn += bf2f_lo(pn) * h0 + bf2f_hi(pn) * h1;
            }
        }
        // reduce across the 8 threads of this column (xor masks stay in-octet)
#pragma unroll
        for (int m = 4; m >= 1; m >>= 1) {
            dr += __shfl_xor(dr, m, 32);
            dz += __shfl_xor(dz, m, 32);
            dn += __shfl_xor(dn, m, 32);
        }

        if (sub == 0) {
            const size_t idx = (size_t)t * HDIM + col;
            const float hprev = sh[col];
            const float r = 1.f / (1.f + __expf(-(xr[idx] + dr)));
            const float z = 1.f / (1.f + __expf(-(xz[idx] + dz)));
            const float n = tanhf(xn_io[idx] + r * (dn + bn));  // read xn before overwrite
            const float hnew = (1.f - z) * hprev + z * n;
            hout[col]  = hnew;
            xn_io[idx] = hnew;        // d_out[t] = h_t
        }

        // grid barrier: monotonic counter, target = (t+1) * NWG
        __syncthreads();
        if (tid == 0) {
            __threadfence();          // release h writes
            atomicAdd(ctr, 1u);
            const unsigned target = (unsigned)(t + 1) * NWG;
            while (__hip_atomic_load(ctr, __ATOMIC_ACQUIRE,
                                     __HIP_MEMORY_SCOPE_AGENT) < target) {
                __builtin_amdgcn_s_sleep(2);
            }
        }
        __syncthreads();
        __threadfence();              // acquire other blocks' h writes
    }
}

// ---------------------------------------------------------------------------
// Host entry. Inputs (setup_inputs order):
//  0 inputs[T,K] 1 W_hr 2 W_hz 3 W_hn 4 b_hn 5 W_ir 6 b_ir 7 W_iz 8 b_iz
//  9 W_in 10 b_in
// ---------------------------------------------------------------------------
extern "C" void kernel_launch(void* const* d_in, const int* in_sizes, int n_in,
                              void* d_out, int out_size, void* d_ws, size_t ws_size,
                              hipStream_t stream) {
    (void)in_sizes; (void)n_in; (void)out_size; (void)ws_size;

    const float* X    = (const float*)d_in[0];
    const float* W_hr = (const float*)d_in[1];
    const float* W_hz = (const float*)d_in[2];
    const float* W_hn = (const float*)d_in[3];
    const float* b_hn = (const float*)d_in[4];
    const float* W_ir = (const float*)d_in[5];
    const float* b_ir = (const float*)d_in[6];
    const float* W_iz = (const float*)d_in[7];
    const float* b_iz = (const float*)d_in[8];
    const float* W_in = (const float*)d_in[9];
    const float* b_in = (const float*)d_in[10];
    float* out = (float*)d_out;

    // workspace layout
    char* ws = (char*)d_ws;
    const size_t xbf_bytes = (size_t)T_LEN * KDIM * 2;        // 16 MB
    const size_t w_bytes   = (size_t)HDIM * HDIM * 2;         // 8 MB each
    const size_t xg_bytes  = (size_t)T_LEN * HDIM * 4;        // 32 MB each
    unsigned short* Xbf  = (unsigned short*)(ws);
    unsigned short* WTir = (unsigned short*)(ws + xbf_bytes + 0 * w_bytes);
    unsigned short* WTiz = (unsigned short*)(ws + xbf_bytes + 1 * w_bytes);
    unsigned short* WTin = (unsigned short*)(ws + xbf_bytes + 2 * w_bytes);
    unsigned short* WThr = (unsigned short*)(ws + xbf_bytes + 3 * w_bytes);
    unsigned short* WThz = (unsigned short*)(ws + xbf_bytes + 4 * w_bytes);
    unsigned short* WThn = (unsigned short*)(ws + xbf_bytes + 5 * w_bytes);
    char* after_w = ws + xbf_bytes + 6 * w_bytes;
    float* xr_buf = (float*)(after_w);
    float* xz_buf = (float*)(after_w + xg_bytes);
    float* hbuf   = (float*)(after_w + 2 * xg_bytes);
    unsigned* ctr = (unsigned*)(after_w + 2 * xg_bytes + 2 * HDIM * 4);

    // 1) init barrier counter + h0 = 0
    init_state<<<1, 256, 0, stream>>>(hbuf, ctr);

    // 2) convert inputs to bf16
    convert_bf16<<<2048, 256, 0, stream>>>(X, Xbf, T_LEN * KDIM);

    // 3) transpose+convert all six weight matrices (column-contiguous bf16)
    dim3 tb(256), tg(HDIM / 32, HDIM / 32);
    transpose_bf16<<<tg, tb, 0, stream>>>(W_ir, WTir, HDIM);
    transpose_bf16<<<tg, tb, 0, stream>>>(W_iz, WTiz, HDIM);
    transpose_bf16<<<tg, tb, 0, stream>>>(W_in, WTin, HDIM);
    transpose_bf16<<<tg, tb, 0, stream>>>(W_hr, WThr, HDIM);
    transpose_bf16<<<tg, tb, 0, stream>>>(W_hz, WThz, HDIM);
    transpose_bf16<<<tg, tb, 0, stream>>>(W_hn, WThn, HDIM);

    // 4) input projections with WMMA: xr, xz -> ws; xn -> d_out
    const int tiles  = (T_LEN / 16) * (HDIM / 16);  // 32768
    const int blocks = tiles / 8;                   // 8 waves per block
    gemm_wmma_bf16<<<blocks, 256, 0, stream>>>(Xbf, WTir, b_ir, xr_buf, T_LEN, KDIM, HDIM);
    gemm_wmma_bf16<<<blocks, 256, 0, stream>>>(Xbf, WTiz, b_iz, xz_buf, T_LEN, KDIM, HDIM);
    gemm_wmma_bf16<<<blocks, 256, 0, stream>>>(Xbf, WTin, b_in, out,    T_LEN, KDIM, HDIM);

    // 5) sequential GRU scan (persistent, grid-synchronized)
    gru_scan<<<NWG, 256, 0, stream>>>(WThr, WThz, WThn, b_hn,
                                      xr_buf, xz_buf, out, hbuf, ctr);
}
